// FitnessHead_57415122813283
// MI455X (gfx1250) — matrix operand order repaired and checked by reading, not compiled, not run on hardware
//
#include <hip/hip_runtime.h>
#include <hip/hip_bf16.h>

// Problem constants (b=1, m=128, n=256, c=22)
#define M_DIM   128
#define N_POS   256
#define C_CLS   22
#define KDIM    (N_POS * C_CLS)   // 5632 = flattened (j,d)
#define NDIM    (N_POS * C_CLS)   // 5632 = flattened (i,c)
#define KT_N    (KDIM / 32)       // 176 K-tiles of 32
#define NT_N    (NDIM / 16)       // 352 N-tiles of 16
#define MT_N    (M_DIM / 16)      // 8 M-tiles of 16
#define EIJ_I_STRIDE (N_POS * C_CLS * C_CLS)  // 123904 floats per i
#define EIJ_J_STRIDE (C_CLS * C_CLS)          // 484 floats per j

typedef __attribute__((ext_vector_type(16))) _Float16 v16h;
typedef __attribute__((ext_vector_type(8)))  float    v8f;

union FragAB { uint4 u[2]; v16h v; };
union H2U    { _Float16 h[2]; unsigned int u; };

// ---------------------------------------------------------------------------
// Kernel 1: build A in WMMA 16-bit A-fragment layout (ISA 7.12.2).
// A[m, k=(j*22+d)] = (variant[m,j]==d) ? mask_vec[d] : 0   (f16, exact here)
// Layout: afrag[((mt*KT_N + kt)*32 + lane)*8 + v]  -> per-lane 32B contiguous.
// Lane<16: M=lane,   v0..3 -> K=2v,2v+1 ; v4..7 -> K=16+2(v-4)..
// Lane>=16: M=lane-16, same pattern shifted by K+=8 (per ISA table).
// ---------------------------------------------------------------------------
__global__ void build_a_frags(const int* __restrict__ variant,
                              const float* __restrict__ mask_vec,
                              unsigned int* __restrict__ afrag) {
  const int kt   = blockIdx.x % KT_N;
  const int mt   = blockIdx.x / KT_N;
  const int lane = threadIdx.x;            // 0..31
  const int m    = mt * 16 + (lane & 15);
  const int khalf = (lane & 16) ? 8 : 0;

  unsigned int out[8];
#pragma unroll
  for (int v = 0; v < 8; ++v) {
    const int klocal = ((v < 4) ? (2 * v) : (8 + 2 * v)) + khalf;
    H2U pack;
#pragma unroll
    for (int h = 0; h < 2; ++h) {
      const int k = kt * 32 + klocal + h;
      const int j = k / C_CLS;
      const int d = k - C_CLS * j;
      const float val = (variant[m * N_POS + j] == d) ? mask_vec[d] : 0.0f;
      pack.h[h] = (_Float16)val;
    }
    out[v] = pack.u;
  }
  unsigned int* dst = afrag + ((size_t)(mt * KT_N + kt) * 32 + lane) * 8;
  *(uint4*)(dst + 0) = *(uint4*)(out + 0);
  *(uint4*)(dst + 4) = *(uint4*)(out + 4);
}

// ---------------------------------------------------------------------------
// Kernel 2: hi = A x B via v_wmma_f32_16x16x32_f16, fold in ei, store motifs.
// One workgroup (8 waves) per 16-wide N-tile; wave w owns M-tile w (all 128 m).
// B tile (32K x 16N, f16) staged in LDS in fragment-ready layout, double buf.
// eij is streamed from HBM exactly once (~127 MB -> memory-bound optimum).
// ---------------------------------------------------------------------------
__global__ __launch_bounds__(256) void potts_gemm(
    const float* __restrict__ eij,
    const float* __restrict__ ei,
    const unsigned int* __restrict__ afrag,
    float* __restrict__ motifs) {
  __shared__ alignas(16) unsigned int ldsB[2][256];  // [buf][lane*8 + v], 2KB

  const int nt   = blockIdx.x;       // 0..351
  const int tid  = threadIdx.x;      // 0..255
  const int wave = tid >> 5;         // = M-tile index, 0..7
  const int lane = tid & 31;

  // --- B-tile writer mapping: thread t fills LDS dword t = wlane*8 + wv ---
  const int wlane = tid >> 3;                       // fragment lane 0..31
  const int wv    = tid & 7;                        // fragment vgpr 0..7
  const int wn    = wlane & 15;                     // N column in tile
  const int wk0   = ((wv < 4) ? (2 * wv) : (8 + 2 * wv)) + ((wlane & 16) ? 8 : 0); // first K of the pair
  const int ng_w  = nt * 16 + wn;                   // global flat (i,c)
  const int wi    = ng_w / C_CLS;
  const int wc    = ng_w - C_CLS * wi;
  const float* ebase = eij + (size_t)wi * EIJ_I_STRIDE + wc * C_CLS;

  v8f acc = {};

  for (int kt = 0; kt < KT_N; ++kt) {
    const int buf = kt & 1;

    // cooperative f32->f16 load of the 32x16 B tile (one dword per thread)
    H2U pack;
#pragma unroll
    for (int h = 0; h < 2; ++h) {
      const int k = kt * 32 + wk0 + h;
      const int j = k / C_CLS;
      const int d = k - C_CLS * j;
      pack.h[h] = (_Float16)ebase[j * EIJ_J_STRIDE + d];
    }
    ldsB[buf][tid] = pack.u;

    __syncthreads();

    FragAB a, b;
    const uint4* ap = (const uint4*)(afrag + ((size_t)(wave * KT_N + kt) * 32 + lane) * 8);
    a.u[0] = ap[0];
    a.u[1] = ap[1];
    const uint4* bp = (const uint4*)(&ldsB[buf][lane * 8]);
    b.u[0] = bp[0];
    b.u[1] = bp[1];

    acc = __builtin_amdgcn_wmma_f32_16x16x32_f16(
        /*neg_a=*/false, a.v, /*neg_b=*/false, b.v,
        /*c_mod=*/(short)0, acc, /*reuse_a=*/false, /*reuse_b=*/false);
  }

  // Epilogue: C/D layout (ISA): lane<16 -> N=lane, M=vgpr r ; lane>=16 -> M=8+r
  const int n  = lane & 15;
  const int ng = nt * 16 + n;          // flat (i,c) == flat index into ei
  const float e = ei[ng];
  const int mbase = wave * 16 + ((lane & 16) ? 8 : 0);
#pragma unroll
  for (int r = 0; r < 8; ++r) {
    motifs[(size_t)(mbase + r) * NDIM + ng] = acc[r] + e;
  }
}

// ---------------------------------------------------------------------------
// Kernel 3: logits[m,i] = motifs[m,i,variant[m,i]]
// ---------------------------------------------------------------------------
__global__ void logits_kernel(const float* __restrict__ motifs,
                              const int* __restrict__ variant,
                              float* __restrict__ logits) {
  const int idx = blockIdx.x * 256 + threadIdx.x;   // 0..32767
  if (idx < M_DIM * N_POS) {
    const int m = idx >> 8;
    const int i = idx & (N_POS - 1);
    const int d = variant[idx];
    logits[idx] = motifs[(size_t)m * NDIM + i * C_CLS + d];
  }
}

// ---------------------------------------------------------------------------
// Kernel 4: variant_logit[m] = sigma * sum_i (logits[m,i]-logits[0,i]) * vm
//           vm = variant_mask[m,i] * variant_mask[0,i]
// ---------------------------------------------------------------------------
__global__ void pool_kernel(const float* __restrict__ logits,
                            const float* __restrict__ vmask,
                            const float* __restrict__ sigma,
                            float* __restrict__ out) {
  __shared__ float red[256];
  const int m = blockIdx.x;
  const int i = threadIdx.x;     // 256 threads = one per position
  const float v = (logits[m * N_POS + i] - logits[i]) *
                  vmask[m * N_POS + i] * vmask[i];
  red[i] = v;
  __syncthreads();
#pragma unroll
  for (int s = 128; s > 0; s >>= 1) {
    if (i < s) red[i] += red[i + s];
    __syncthreads();
  }
  if (i == 0) out[m] = red[0] * sigma[0];
}

// ---------------------------------------------------------------------------
extern "C" void kernel_launch(void* const* d_in, const int* in_sizes, int n_in,
                              void* d_out, int out_size, void* d_ws, size_t ws_size,
                              hipStream_t stream) {
  const int*   variant  = (const int*)d_in[0];
  const float* vmask    = (const float*)d_in[1];
  const float* eij      = (const float*)d_in[2];
  const float* ei       = (const float*)d_in[3];
  const float* mask_vec = (const float*)d_in[4];
  const float* sigma    = (const float*)d_in[5];

  float* motifs = (float*)d_out;                    // 128*5632 = 720896
  float* logits = motifs + (size_t)M_DIM * NDIM;    // +32768
  float* vlogit = logits + (size_t)M_DIM * N_POS;   // +128

  // Workspace: A fragments, 8*176*32*8 dwords = 1,441,792 bytes
  unsigned int* afrag = (unsigned int*)d_ws;

  build_a_frags<<<MT_N * KT_N, 32, 0, stream>>>(variant, mask_vec, afrag);
  potts_gemm<<<NT_N, 256, 0, stream>>>(eij, ei, afrag, motifs);
  logits_kernel<<<(M_DIM * N_POS + 255) / 256, 256, 0, stream>>>(motifs, variant, logits);
  pool_kernel<<<M_DIM, 256, 0, stream>>>(logits, vmask, sigma, vlogit);
}
